// MultiHeadAttentionWithRoPE_64364379898329
// MI455X (gfx1250) — matrix operand order repaired and checked by reading, not compiled, not run on hardware
//
#include <hip/hip_runtime.h>
#include <hip/hip_bf16.h>

// ---------------------------------------------------------------------------
// MHA + RoPE for MI455X (gfx1250), f16 WMMA with f32 accumulation.
// L=4096, D=1024, H=16, dh=64.
// Flash attention computes S^T = K Q^T and O^T = V^T P^T so the softmax
// dimension is register-local per lane (one query per lane).
// ---------------------------------------------------------------------------

typedef _Float16 v16h __attribute__((ext_vector_type(16)));
typedef float    v8f  __attribute__((ext_vector_type(8)));

#define L_SEQ  4096
#define D_MOD  1024
#define NHEAD  16
#define DHEAD  64

// ---------------- fp32 -> fp16 conversion ----------------------------------
__global__ void cast_f32_to_f16(const float* __restrict__ in,
                                _Float16* __restrict__ out, int n) {
    int i = blockIdx.x * blockDim.x + threadIdx.x;
    if (i < n) out[i] = (_Float16)in[i];
}

// ---------------- RoPE (in-place on f16 Q,K); Q also scaled by 1/sqrt(dh) --
__global__ void rope_kernel(_Float16* __restrict__ q, _Float16* __restrict__ k,
                            const int* __restrict__ pos) {
    int idx = blockIdx.x * blockDim.x + threadIdx.x;   // L * D/2 pairs
    if (idx >= L_SEQ * (D_MOD / 2)) return;
    int l   = idx >> 9;           // D/2 = 512 pairs per row
    int j   = idx & 511;
    int col = 2 * j;
    int fh  = col & (DHEAD - 1);  // even index within the head
    // freq = theta^(-fh/dh) = exp(-ln(10000)/64 * fh)
    float freq = __expf(-0.14391156516f * (float)fh);
    float ang  = (float)pos[l] * freq;
    float s, c;
    __sincosf(ang, &s, &c);
    int base = l * D_MOD + col;
    float q1 = (float)q[base], q2 = (float)q[base + 1];
    q[base]     = (_Float16)(0.125f * (q1 * c - q2 * s));
    q[base + 1] = (_Float16)(0.125f * (q1 * s + q2 * c));
    float k1 = (float)k[base], k2 = (float)k[base + 1];
    k[base]     = (_Float16)(k1 * c - k2 * s);
    k[base + 1] = (_Float16)(k1 * s + k2 * c);
}

// ---------------- V transpose: VT[h*64+d][l] = V[l][h*64+d] ----------------
__global__ void transpose_v(const _Float16* __restrict__ v,
                            _Float16* __restrict__ vt) {
    int idx = blockIdx.x * blockDim.x + threadIdx.x;   // 1024*4096
    if (idx >= D_MOD * L_SEQ) return;
    int l  = idx & (L_SEQ - 1);
    int hd = idx >> 12;
    vt[idx] = v[l * D_MOD + hd];
}

// ---------------- fragment loaders -----------------------------------------
// A-matrix 16x32 f16 layout: lane holds row m=L%16; lanes 0-15 take K=0..15,
// lanes 16-31 take K=16..31 (contiguous, matching B).
__device__ __forceinline__ v16h load_afrag(const _Float16* rowptr, int k, int hf) {
    v16h r;
    ((uint4*)&r)[0] = *(const uint4*)(rowptr + k + hf * 16);
    ((uint4*)&r)[1] = *(const uint4*)(rowptr + k + 8 + hf * 16);
    return r;
}
// B-matrix 32x16 f16 layout: lane holds column n=L%16; lanes 0-15 K=0..15,
// lanes 16-31 K=16..31 (contiguous 16 halves).
__device__ __forceinline__ v16h load_bfrag(const _Float16* colptr, int k, int hf) {
    v16h r;
    const uint4* p = (const uint4*)(colptr + k + hf * 16);
    ((uint4*)&r)[0] = p[0];
    ((uint4*)&r)[1] = p[1];
    return r;
}

__device__ __forceinline__ unsigned pkh2(float a, float b) {
    auto h = __builtin_amdgcn_cvt_pkrtz(a, b);   // v_cvt_pk_rtz_f16_f32, __fp16x2
    return __builtin_bit_cast(unsigned, h);
}

// ---------------- GEMM: out[m,n] = sum_k A[m,k] * W[n,k] -------------------
// A: 4096x1024 f16 row-major, W: 1024x1024 f16 row-major.
// Block = 128 threads (4 waves); block tile 128x128; wave tile 64x64.
template <typename OT>
__global__ void gemm_wmma(const _Float16* __restrict__ A,
                          const _Float16* __restrict__ W,
                          OT* __restrict__ out) {
    int lane = threadIdx.x & 31;
    int warp = threadIdx.x >> 5;
    int hf   = lane >> 4;
    int l16  = lane & 15;
    int nBlk = blockIdx.x * 128 + (warp & 1) * 64;
    int mBlk = blockIdx.y * 128 + (warp >> 1) * 64;

    const _Float16* arow[4];
    const _Float16* bcol[4];
#pragma unroll
    for (int i = 0; i < 4; ++i) arow[i] = A + (size_t)(mBlk + i * 16 + l16) * D_MOD;
#pragma unroll
    for (int j = 0; j < 4; ++j) bcol[j] = W + (size_t)(nBlk + j * 16 + l16) * D_MOD;

    v8f c[4][4];
#pragma unroll
    for (int i = 0; i < 4; ++i)
#pragma unroll
        for (int j = 0; j < 4; ++j) c[i][j] = 0.0f;

    for (int k = 0; k < D_MOD; k += 32) {
        v16h a[4], b[4];
#pragma unroll
        for (int i = 0; i < 4; ++i) {
            a[i] = load_afrag(arow[i], k, hf);
            __builtin_prefetch(arow[i] + k + 64, 0, 0);   // global_prefetch
        }
#pragma unroll
        for (int j = 0; j < 4; ++j) {
            b[j] = load_bfrag(bcol[j], k, hf);
            __builtin_prefetch(bcol[j] + k + 64, 0, 0);
        }
#pragma unroll
        for (int i = 0; i < 4; ++i)
#pragma unroll
            for (int j = 0; j < 4; ++j)
                c[i][j] = __builtin_amdgcn_wmma_f32_16x16x32_f16(
                    false, a[i], false, b[j], (short)0, c[i][j], false, false);
    }

#pragma unroll
    for (int i = 0; i < 4; ++i)
#pragma unroll
        for (int j = 0; j < 4; ++j)
#pragma unroll
            for (int r = 0; r < 8; ++r) {
                int row = mBlk + i * 16 + r + 8 * hf;
                int col = nBlk + j * 16 + l16;
                out[(size_t)row * D_MOD + col] = (OT)c[i][j][r];
            }
}

// ---------------- Flash attention (transposed dataflow) --------------------
// One wave per (head, 16-query block); key blocks of 32.
// Lane owns query q = lane%16; running max/sum are per-lane scalars.
__global__ void flash_attn(const _Float16* __restrict__ Q,
                           const _Float16* __restrict__ K,
                           const _Float16* __restrict__ VT,
                           _Float16* __restrict__ Oa) {
    int lane = threadIdx.x & 31;
    int warp = threadIdx.x >> 5;
    int hf   = lane >> 4;
    int l16  = lane & 15;
    int wid  = blockIdx.x * 4 + warp;    // 0..4095
    int h    = wid >> 8;                 // head
    int qb   = wid & 255;                // query block (16 rows)

    // Q^T as B-operand: column q = this lane's query row (contiguous over d).
    const _Float16* qcol = Q + (size_t)(qb * 16 + l16) * D_MOD + h * DHEAD;
    v16h qf0 = load_bfrag(qcol, 0, hf);    // d = 0..31
    v16h qf1 = load_bfrag(qcol, 32, hf);   // d = 32..63

    v8f o[4];                              // O^T tiles: d = t*16 + r + 8*hf
#pragma unroll
    for (int t = 0; t < 4; ++t) o[t] = 0.0f;
    float mrun = -1e30f, lrun = 0.0f;

    for (int kb = 0; kb < L_SEQ; kb += 32) {
        // ---- S^T = K Q^T : (32 keys) x (16 queries), two 16-key C tiles ----
        const _Float16* krow0 = K + (size_t)(kb + l16) * D_MOD + h * DHEAD;
        const _Float16* krow1 = krow0 + (size_t)16 * D_MOD;
        v8f st0 = 0.0f, st1 = 0.0f;
        {
            v16h ka = load_afrag(krow0, 0, hf);
            st0 = __builtin_amdgcn_wmma_f32_16x16x32_f16(false, ka, false, qf0,
                                                         (short)0, st0, false, false);
            ka = load_afrag(krow0, 32, hf);
            st0 = __builtin_amdgcn_wmma_f32_16x16x32_f16(false, ka, false, qf1,
                                                         (short)0, st0, false, false);
            ka = load_afrag(krow1, 0, hf);
            st1 = __builtin_amdgcn_wmma_f32_16x16x32_f16(false, ka, false, qf0,
                                                         (short)0, st1, false, false);
            ka = load_afrag(krow1, 32, hf);
            st1 = __builtin_amdgcn_wmma_f32_16x16x32_f16(false, ka, false, qf1,
                                                         (short)0, st1, false, false);
        }
        // Lane holds 16 scores for query l16: keys kb + r + 8*hf (st0)
        // and kb + 16 + r + 8*hf (st1). Other 16 keys live in lane^16.

        // ---- online softmax: register-local + one half-swap shuffle --------
        float mloc = fmaxf(st0[0], st1[0]);
#pragma unroll
        for (int r = 1; r < 8; ++r) mloc = fmaxf(mloc, fmaxf(st0[r], st1[r]));
        mloc = fmaxf(mloc, __shfl_xor(mloc, 16, 32));
        float mnew  = fmaxf(mrun, mloc);
        float alpha = __expf(mrun - mnew);
        float p0[8], p1[8], sloc = 0.0f;
#pragma unroll
        for (int r = 0; r < 8; ++r) {
            p0[r] = __expf(st0[r] - mnew);
            p1[r] = __expf(st1[r] - mnew);
            sloc += p0[r] + p1[r];
        }
        sloc += __shfl_xor(sloc, 16, 32);
        lrun = lrun * alpha + sloc;
        mrun = mnew;

        // ---- build P^T B-fragment: pack f16 pairs, swap lane halves -------
        // Lane half owns: u = st0 pack (keys kb+8*hf+0..7 pairs),
        //                 w = st1 pack (keys kb+16+8*hf+0..7 pairs).
        // half 0 needs partner's u (keys 8..15); half 1 needs partner's w
        // (keys 16..23). Send exactly what the partner needs: 4 shuffles.
        unsigned u[4], w[4], rx[4];
#pragma unroll
        for (int v = 0; v < 4; ++v) {
            u[v] = pkh2(p0[2 * v], p0[2 * v + 1]);
            w[v] = pkh2(p1[2 * v], p1[2 * v + 1]);
        }
#pragma unroll
        for (int v = 0; v < 4; ++v) {
            unsigned payload = hf ? u[v] : w[v];       // what my partner needs
            rx[v] = __shfl_xor(payload, 16, 32);       // what I need
        }
        v16h pb;
        unsigned* pd = (unsigned*)&pb;
#pragma unroll
        for (int v = 0; v < 4; ++v) {
            // lanes 0-15 (K=0..15): own u (0..7) then partner u (8..15)
            // lanes 16-31 (K=16..31): partner w (16..23) then own w (24..31)
            pd[v]     = hf ? rx[v] : u[v];
            pd[4 + v] = hf ? w[v] : rx[v];
        }

        // ---- rescale O^T and accumulate V^T P^T ---------------------------
#pragma unroll
        for (int t = 0; t < 4; ++t) o[t] = o[t] * alpha;
#pragma unroll
        for (int t = 0; t < 4; ++t) {
            const _Float16* vrow =
                VT + (size_t)(h * DHEAD + t * 16 + l16) * L_SEQ + kb;
            v16h va = load_afrag(vrow, 0, hf);
            o[t] = __builtin_amdgcn_wmma_f32_16x16x32_f16(false, va, false, pb,
                                                          (short)0, o[t], false, false);
        }
    }

    // ---- normalize, pack, write: per lane 8 contiguous f16 per tile -------
    float inv = 1.0f / lrun;
    int row = qb * 16 + l16;                       // this lane's query
    _Float16* obase = Oa + (size_t)row * D_MOD + h * DHEAD;
#pragma unroll
    for (int t = 0; t < 4; ++t) {
        uint4 pk;
        pk.x = pkh2(o[t][0] * inv, o[t][1] * inv);
        pk.y = pkh2(o[t][2] * inv, o[t][3] * inv);
        pk.z = pkh2(o[t][4] * inv, o[t][5] * inv);
        pk.w = pkh2(o[t][6] * inv, o[t][7] * inv);
        *(uint4*)(obase + t * 16 + 8 * hf) = pk;   // cols t*16+8*hf .. +7
    }
}

// ---------------------------------------------------------------------------
extern "C" void kernel_launch(void* const* d_in, const int* in_sizes, int n_in,
                              void* d_out, int out_size, void* d_ws, size_t ws_size,
                              hipStream_t stream) {
    const float* x  = (const float*)d_in[0];
    const int*   ps = (const int*)d_in[1];
    const float* wq = (const float*)d_in[2];
    const float* wk = (const float*)d_in[3];
    const float* wv = (const float*)d_in[4];
    const float* wo = (const float*)d_in[5];
    float* out = (float*)d_out;

    const size_t XN = (size_t)L_SEQ * D_MOD;       // 4 Mi elems
    const size_t WN = (size_t)D_MOD * D_MOD;       // 1 Mi elems
    char* ws = (char*)d_ws;
    _Float16* xh  = (_Float16*)(ws);
    _Float16* qh  = (_Float16*)(ws + 1 * XN * 2);
    _Float16* kh  = (_Float16*)(ws + 2 * XN * 2);
    _Float16* vh  = (_Float16*)(ws + 3 * XN * 2);
    _Float16* vth = (_Float16*)(ws + 4 * XN * 2);
    _Float16* ah  = (_Float16*)(ws + 5 * XN * 2);
    _Float16* wqh = (_Float16*)(ws + 6 * XN * 2);
    _Float16* wkh = (_Float16*)(ws + 6 * XN * 2 + 1 * WN * 2);
    _Float16* wvh = (_Float16*)(ws + 6 * XN * 2 + 2 * WN * 2);
    _Float16* woh = (_Float16*)(ws + 6 * XN * 2 + 3 * WN * 2);

    // 1) fp32 -> fp16
    cast_f32_to_f16<<<(int)((XN + 255) / 256), 256, 0, stream>>>(x, xh, (int)XN);
    cast_f32_to_f16<<<(int)((WN + 255) / 256), 256, 0, stream>>>(wq, wqh, (int)WN);
    cast_f32_to_f16<<<(int)((WN + 255) / 256), 256, 0, stream>>>(wk, wkh, (int)WN);
    cast_f32_to_f16<<<(int)((WN + 255) / 256), 256, 0, stream>>>(wv, wvh, (int)WN);
    cast_f32_to_f16<<<(int)((WN + 255) / 256), 256, 0, stream>>>(wo, woh, (int)WN);

    // 2) Q,K,V projections (x @ W^T), f16 out
    dim3 ggrid(D_MOD / 128, L_SEQ / 128);
    gemm_wmma<_Float16><<<ggrid, 128, 0, stream>>>(xh, wqh, qh);
    gemm_wmma<_Float16><<<ggrid, 128, 0, stream>>>(xh, wkh, kh);
    gemm_wmma<_Float16><<<ggrid, 128, 0, stream>>>(xh, wvh, vh);

    // 3) RoPE on Q,K (Q gets 1/sqrt(dh) folded in)
    int npairs = L_SEQ * (D_MOD / 2);
    rope_kernel<<<(npairs + 255) / 256, 256, 0, stream>>>(qh, kh, ps);

    // 4) transpose V per head for V^T A-fragment loads
    transpose_v<<<(int)((XN + 255) / 256), 256, 0, stream>>>(vh, vth);

    // 5) flash attention: 16 heads * 256 query blocks = 4096 waves
    flash_attn<<<(NHEAD * (L_SEQ / 16)) / 4, 128, 0, stream>>>(qh, kh, vth, ah);

    // 6) output projection attn @ Wo^T, f32 out
    gemm_wmma<float><<<ggrid, 128, 0, stream>>>(ah, woh, out);
}
